// Galerkin_attn_20486994002476
// MI455X (gfx1250) — matrix operand order repaired and checked by reading, not compiled, not run on hardware
//
#include <hip/hip_runtime.h>
#include <hip/hip_bf16.h>
#include <math.h>

// MI455X / gfx1250, wave32. bf16 WMMA 16x16x32, f32 accumulate.
typedef __attribute__((ext_vector_type(16))) __bf16 v16bf;
typedef __attribute__((ext_vector_type(8)))  float  v8f;

// TDM descriptor vector types (probe-verified builtin signatures)
typedef __attribute__((ext_vector_type(4))) unsigned tdm_v4u;
typedef __attribute__((ext_vector_type(8))) int      tdm_v8i;
typedef __attribute__((ext_vector_type(4))) int      tdm_v4i;

#if __has_builtin(__builtin_amdgcn_tensor_load_to_lds)
#define HAVE_TDM 1
#if __has_include(<hip/amd_detail/amd_gfx1250_TDM.h>)
#define TDM_6ARG 1   // amdgpu-toolchain / therock-10.0 headers: 6-arg builtin
#endif
#endif

#define BATCH  8
#define CHN    128
#define NPIX   65536      // 256*256
#define NHEADS 8
#define EPSLN  1e-5f

// workspace layout (bytes); total 491,520 B
#define WS_QW   0          // gathered q weights  128x128 bf16
#define WS_KVW  32768      // gathered k|v weights 256x128 bf16
#define WS_O1W  98304      // o1 weights 128x128 bf16
#define WS_O2W  131072     // o2 weights 128x128 bf16
#define WS_KV   163840     // kv accumulator B*8*16*16 f32 (atomics)
#define WS_KVT  229376     // block-diag KV^T, per batch 128x128 bf16

__device__ __forceinline__ __bf16 f2bf(float f) {
  union { float f; unsigned u; } a; a.f = f;
  unsigned r = a.u + 0x7fffu + ((a.u >> 16) & 1u);   // RNE
  union { unsigned short s; __bf16 b; } o; o.s = (unsigned short)(r >> 16);
  return o.b;
}

__device__ __forceinline__ float gelu_exact(float x) {
  return 0.5f * x * (1.0f + erff(x * 0.70710678118654752f));
}

// A fragment (16x32 bf16) from row-major [row][ld] at (row_base, kbase).
// Layout: lane l holds M=l%16; element s: K = 16*(l/16) + s (contiguous, matches load_b)
__device__ __forceinline__ v16bf load_a(const __bf16* W, int row_base, int ld, int kbase, int lane) {
  const unsigned short* p = (const unsigned short*)(W + (row_base + (lane & 15)) * ld + kbase);
  int h8 = (lane >> 4) << 3;
  union { v16bf v; unsigned u[8]; } f;
#pragma unroll
  for (int r = 0; r < 4; ++r) f.u[r]     = *(const unsigned*)(p + 2 * h8 + 2 * r);
#pragma unroll
  for (int r = 0; r < 4; ++r) f.u[4 + r] = *(const unsigned*)(p + 8 + 2 * h8 + 2 * r);
  return f.v;
}

// B fragment (32x16 bf16) from row-major [n][ld] (row = N, col = K).
// Layout: lane l holds N=l%16, K = kbase + 16*(l/16) + j (16 contiguous) -> 2x 16B LDS loads
__device__ __forceinline__ v16bf load_b(const __bf16* Xp, int ld, int nbase, int kbase, int lane) {
  const uint4* p = (const uint4*)(Xp + (nbase + (lane & 15)) * ld + kbase + ((lane >> 4) << 4));
  union { v16bf v; uint4 q[2]; } f;
  f.q[0] = p[0]; f.q[1] = p[1];
  return f.v;
}

__device__ __forceinline__ v8f wmma_bf16(v16bf a, v16bf b, v8f c) {
  return __builtin_amdgcn_wmma_f32_16x16x32_bf16(false, a, false, b, (short)0, c, false, false);
}

// ---------------- kernel 0: weight prep (bf16, gathered rows) + zero kv accumulator ----------
__global__ void prep_kernel(const float* qkv_w, const float* o1_w, const float* o2_w, char* ws) {
  int idx = blockIdx.x * 256 + threadIdx.x;            // 32768 threads
  __bf16* qw  = (__bf16*)(ws + WS_QW);
  __bf16* kvw = (__bf16*)(ws + WS_KVW);
  __bf16* o1  = (__bf16*)(ws + WS_O1W);
  __bf16* o2  = (__bf16*)(ws + WS_O2W);
  float*  kv  = (float*)(ws + WS_KV);
  if (idx < 16384) {
    int co = idx >> 7, ci = idx & 127;
    int h = co >> 4, d = co & 15;
    qw[idx] = f2bf(qkv_w[(48 * h + d) * CHN + ci]);    // q rows: 48h..48h+15
    o1[idx] = f2bf(o1_w[idx]);
    o2[idx] = f2bf(o2_w[idx]);
    kv[idx] = 0.0f;                                    // 8*8*16*16 floats
  }
  if (idx < 32768) {
    int rr = idx >> 7, ci = idx & 127;
    int h = rr >> 5, t = rr & 31;
    int src = (t < 16) ? (48 * h + 16 + t) : (48 * h + 32 + (t - 16));  // k then v rows per head
    kvw[idx] = f2bf(qkv_w[src * CHN + ci]);
  }
}

// ---------------- kernel 1: k,v GEMM + LayerNorm + kv outer-product accumulation -------------
// grid = BATCH * 512 (128-pixel tiles), block = 256 (8 waves)
__global__ void __launch_bounds__(256) qkv_kv_kernel(const float* x,
    const float* kln_w, const float* kln_b, const float* vln_w, const float* vln_b, char* ws) {
  extern __shared__ char smem[];
  float*  kvt = (float*)smem;                 // [128 n][256 rr] f32      131072 B
  __bf16* xb  = (__bf16*)(smem + 131072);     // [128 n][144]  bf16 x     36864 B
  __bf16* kT  = (__bf16*)(smem + 167936);     // [128 hd][144 n] bf16 k   36864 B
  __bf16* vT  = (__bf16*)(smem + 204800);     // [128 he][144 n] bf16 v   36864 B

  int tid = threadIdx.x;
  int b = blockIdx.x >> 9, tile = blockIdx.x & 511;
  long n0 = (long)tile * 128;
  const float* xbase = x + (long)b * CHN * NPIX;

  for (int i = tid; i < CHN * 128; i += 256) {          // stage x tile -> xb[n][c] bf16
    int c = i >> 7, j = i & 127;
    xb[j * 144 + c] = f2bf(xbase[(long)c * NPIX + n0 + j]);
  }
  __syncthreads();

  int wv = tid >> 5, lane = tid & 31;
  int h8 = (lane >> 4) << 3, l15 = lane & 15;
  const __bf16* kvw = (const __bf16*)(ws + WS_KVW);

  v8f zero = {};
  v8f acc[2][8];
#pragma unroll
  for (int a = 0; a < 2; ++a)
#pragma unroll
    for (int n = 0; n < 8; ++n) acc[a][n] = zero;

  for (int kt = 0; kt < 4; ++kt) {                      // k|v rows GEMM: wave w = head w
    int kb = kt * 32;
    v16bf a0 = load_a(kvw, (2 * wv) * 16, CHN, kb, lane);
    v16bf a1 = load_a(kvw, (2 * wv + 1) * 16, CHN, kb, lane);
#pragma unroll
    for (int nt = 0; nt < 8; ++nt) {
      v16bf bf = load_b(xb, 144, nt * 16, kb, lane);
      acc[0][nt] = wmma_bf16(a0, bf, acc[0][nt]);
      acc[1][nt] = wmma_bf16(a1, bf, acc[1][nt]);
    }
  }
#pragma unroll
  for (int ct = 0; ct < 2; ++ct)
#pragma unroll
    for (int nt = 0; nt < 8; ++nt) {
      int rrb = (2 * wv + ct) * 16 + h8;
      int n = nt * 16 + l15;
#pragma unroll
      for (int r = 0; r < 8; ++r) kvt[n * 256 + rrb + r] = acc[ct][nt][r];
    }
  __syncthreads();

  for (int it = tid; it < 128 * NHEADS; it += 256) {    // LN over 16 chans, per (pixel, head)
    int n = it >> 3, h = it & 7;
    const float* kr = kvt + n * 256 + 32 * h;
    const float* vr = kr + 16;
    float ks = 0, kq = 0, vs = 0, vq = 0;
#pragma unroll
    for (int d = 0; d < 16; ++d) {
      float kk = kr[d], vv = vr[d];
      ks += kk; kq += kk * kk; vs += vv; vq += vv * vv;
    }
    float km = ks * (1.f / 16.f);
    float kstd = sqrtf(fmaxf((kq - 16.f * km * km) * (1.f / 15.f), 0.f)) + EPSLN; // unbiased, eps on std
    float vm = vs * (1.f / 16.f);
    float vstd = sqrtf(fmaxf((vq - 16.f * vm * vm) * (1.f / 15.f), 0.f)) + EPSLN;
#pragma unroll
    for (int d = 0; d < 16; ++d) {
      kT[(h * 16 + d) * 144 + n] = f2bf(kln_w[h * 16 + d] * ((kr[d] - km) / kstd) + kln_b[h * 16 + d]);
      vT[(h * 16 + d) * 144 + n] = f2bf(vln_w[h * 16 + d] * ((vr[d] - vm) / vstd) + vln_b[h * 16 + d]);
    }
  }
  __syncthreads();

  v8f kvacc = zero;                                     // kv[d][e] += sum_n k[n][d] v[n][e]
  for (int nb = 0; nb < 4; ++nb) {
    int kb = nb * 32;
    v16bf ak = load_a(kT, wv * 16, 144, kb, lane);                 // A[M=d][K=n]
    v16bf bv = load_b(vT + (wv * 16) * 144, 144, 0, kb, lane);     // B[K=n][N=e]
    kvacc = wmma_bf16(ak, bv, kvacc);
  }
  float* kvg = (float*)(ws + WS_KV) + ((b * NHEADS + wv) * 16) * 16;
#pragma unroll
  for (int r = 0; r < 8; ++r)
    atomicAdd(kvg + (r + h8) * 16 + l15, kvacc[r]);
}

// ---------------- kernel 2: expand kv -> block-diagonal KV^T (bf16), fold 1/N ----------------
__global__ void kvmat_kernel(char* ws) {
  int idx = blockIdx.x * 256 + threadIdx.x;             // BATCH*128*128 = 131072
  int b = idx >> 14, co = (idx >> 7) & 127, ci = idx & 127;
  const float* kv = (const float*)(ws + WS_KV);
  __bf16* kvtb = (__bf16*)(ws + WS_KVT);
  float v = 0.f;
  if ((co >> 4) == (ci >> 4)) {
    int h = co >> 4;
    v = kv[((b * NHEADS + h) * 16 + (ci & 15)) * 16 + (co & 15)] * (1.0f / (float)NPIX);
  }
  kvtb[idx] = f2bf(v);
}

// ---------------- kernel 3: fused q -> o(+x) -> o1+gelu -> o2+x+gelu -------------------------
// grid = BATCH * 256 (256-pixel tiles), block = 256 (8 waves)
__global__ void __launch_bounds__(256) out_kernel(const float* x, float* out, const char* ws) {
  extern __shared__ char smem[];
  float*  xf  = (float*)smem;                 // [128 c][260 n] f32      133120 B
  __bf16* xb  = (__bf16*)(smem + 133120);     // [256 n][144 c] bf16      73728 B
  __bf16* buf = (__bf16*)(smem + 206848);     // ping-pong [n][c] bf16    73728 B

  int tid = threadIdx.x;
  int b = blockIdx.x >> 8, tile = blockIdx.x & 255;
  long n0 = (long)tile * 256;
  const float* xbase = x + (long)b * CHN * NPIX;

#ifdef HAVE_TDM
  // ---- TDM: DMA the fp32 x tile (128 rows x 256 f32, row stride NPIX) into xf with
  //      hardware LDS padding: interval 256 dwords (code 7), pad 4 dwords (code 3) -> stride 260.
  if (tid < 32) {                                        // wave 0 issues one tensor op
    unsigned long long ga = (unsigned long long)(uintptr_t)(xbase + n0);
    unsigned lds_addr = (unsigned)(uintptr_t)xf;         // AS(3) offset in low 32 bits
    tdm_v4u g0 = { 1u,                                   // count=1 valid descriptor
                   lds_addr,
                   (unsigned)ga,
                   (unsigned)((ga >> 32) & 0x01FFFFFFull) | 0x80000000u };   // addr[56:32] | type=2
    tdm_v8i g1 = { (int)0x07D20000,                      // data_size=4B, pad_en, intv=256dw, amt=4dw
                   0,                                    // atomic barrier off / tensor_dim0 lo16 = 0
                   (int)0x00800001,                      // tensor_dim0 hi (65536) | tensor_dim1 = 128
                   (int)0x01000000,                      // tile_dim0 = 256
                   (int)0x00000080,                      // tile_dim1 = 128, tile_dim2 = 0
                   (int)0x00010000,                      // tensor_dim0_stride = 65536
                   0, 0 };
    tdm_v4i gz = { 0, 0, 0, 0 };                         // 2-D tensor: groups 2/3 unused
#ifdef TDM_6ARG
    tdm_v8i gz8 = { 0, 0, 0, 0, 0, 0, 0, 0 };
    __builtin_amdgcn_tensor_load_to_lds(g0, g1, gz, gz, gz8, 0);
#else
    __builtin_amdgcn_tensor_load_to_lds(g0, g1, gz, gz, 0);
#endif
  }
  // concurrent with the DMA: bf16 conversion pass (re-reads hit just-fetched L2/WGP$ lines)
  for (int i = tid; i < CHN * 256; i += 256) {
    int c = i >> 8, j = i & 255;
    xb[j * 144 + c] = f2bf(xbase[(long)c * NPIX + n0 + j]);
  }
  if (tid < 32) __builtin_amdgcn_s_wait_tensorcnt(0);    // fence DMA before barrier release
  __syncthreads();
#else
  for (int i = tid; i < CHN * 256; i += 256) {
    int c = i >> 8, j = i & 255;
    float v = xbase[(long)c * NPIX + n0 + j];
    xf[c * 260 + j] = v;
    xb[j * 144 + c] = f2bf(v);
  }
  __syncthreads();
#endif

  int wv = tid >> 5, lane = tid & 31;
  int h8 = (lane >> 4) << 3, l15 = lane & 15;
  const __bf16* qw   = (const __bf16*)(ws + WS_QW);
  const __bf16* o1w  = (const __bf16*)(ws + WS_O1W);
  const __bf16* o2w  = (const __bf16*)(ws + WS_O2W);
  const __bf16* kvtb = (const __bf16*)(ws + WS_KVT) + b * 16384;

  v8f zero = {};
  v8f acc[16];

  // stage 1: q = qw @ x (recomputed, not stored to HBM)
#pragma unroll
  for (int nt = 0; nt < 16; ++nt) acc[nt] = zero;
  for (int kt = 0; kt < 4; ++kt) {
    int kb = kt * 32;
    v16bf a = load_a(qw, wv * 16, CHN, kb, lane);
#pragma unroll
    for (int nt = 0; nt < 16; ++nt)
      acc[nt] = wmma_bf16(a, load_b(xb, 144, nt * 16, kb, lane), acc[nt]);
  }
#pragma unroll
  for (int nt = 0; nt < 16; ++nt)
#pragma unroll
    for (int r = 0; r < 8; ++r)
      buf[(nt * 16 + l15) * 144 + wv * 16 + h8 + r] = f2bf(acc[nt][r]);
  __syncthreads();

  // stage 2: o = KV^T @ q ; ret = o + x
#pragma unroll
  for (int nt = 0; nt < 16; ++nt) acc[nt] = zero;
  for (int kt = 0; kt < 4; ++kt) {
    int kb = kt * 32;
    v16bf a = load_a(kvtb, wv * 16, CHN, kb, lane);
#pragma unroll
    for (int nt = 0; nt < 16; ++nt)
      acc[nt] = wmma_bf16(a, load_b(buf, 144, nt * 16, kb, lane), acc[nt]);
  }
#pragma unroll
  for (int nt = 0; nt < 16; ++nt)
#pragma unroll
    for (int r = 0; r < 8; ++r)
      acc[nt][r] += xf[(wv * 16 + h8 + r) * 260 + nt * 16 + l15];
  __syncthreads();
#pragma unroll
  for (int nt = 0; nt < 16; ++nt)
#pragma unroll
    for (int r = 0; r < 8; ++r)
      buf[(nt * 16 + l15) * 144 + wv * 16 + h8 + r] = f2bf(acc[nt][r]);
  __syncthreads();

  // stage 3: u = gelu(o1 @ ret)
#pragma unroll
  for (int nt = 0; nt < 16; ++nt) acc[nt] = zero;
  for (int kt = 0; kt < 4; ++kt) {
    int kb = kt * 32;
    v16bf a = load_a(o1w, wv * 16, CHN, kb, lane);
#pragma unroll
    for (int nt = 0; nt < 16; ++nt)
      acc[nt] = wmma_bf16(a, load_b(buf, 144, nt * 16, kb, lane), acc[nt]);
  }
#pragma unroll
  for (int nt = 0; nt < 16; ++nt)
#pragma unroll
    for (int r = 0; r < 8; ++r)
      acc[nt][r] = gelu_exact(acc[nt][r]);
  __syncthreads();
#pragma unroll
  for (int nt = 0; nt < 16; ++nt)
#pragma unroll
    for (int r = 0; r < 8; ++r)
      buf[(nt * 16 + l15) * 144 + wv * 16 + h8 + r] = f2bf(acc[nt][r]);
  __syncthreads();

  // stage 4: out = gelu(o2 @ u + x)
#pragma unroll
  for (int nt = 0; nt < 16; ++nt) acc[nt] = zero;
  for (int kt = 0; kt < 4; ++kt) {
    int kb = kt * 32;
    v16bf a = load_a(o2w, wv * 16, CHN, kb, lane);
#pragma unroll
    for (int nt = 0; nt < 16; ++nt)
      acc[nt] = wmma_bf16(a, load_b(buf, 144, nt * 16, kb, lane), acc[nt]);
  }
#pragma unroll
  for (int nt = 0; nt < 16; ++nt)
#pragma unroll
    for (int r = 0; r < 8; ++r) {
      int c = wv * 16 + h8 + r;
      int n = nt * 16 + l15;
      float v = gelu_exact(acc[nt][r] + xf[c * 260 + n]);
      out[((long)(b * CHN + c)) * NPIX + n0 + n] = v;   // 16 lanes -> 64B contiguous per row
    }
}

extern "C" void kernel_launch(void* const* d_in, const int* in_sizes, int n_in,
                              void* d_out, int out_size, void* d_ws, size_t ws_size,
                              hipStream_t stream) {
  (void)in_sizes; (void)n_in; (void)out_size; (void)ws_size;
  const float* x     = (const float*)d_in[0];
  const float* qkv_w = (const float*)d_in[1];
  const float* o1_w  = (const float*)d_in[2];
  const float* o2_w  = (const float*)d_in[3];
  const float* kln_w = (const float*)d_in[4];
  const float* kln_b = (const float*)d_in[5];
  const float* vln_w = (const float*)d_in[6];
  const float* vln_b = (const float*)d_in[7];
  float* out = (float*)d_out;
  char* ws = (char*)d_ws;   // uses 491,520 bytes

  hipFuncSetAttribute((const void*)qkv_kv_kernel, hipFuncAttributeMaxDynamicSharedMemorySize, 241664);
  hipFuncSetAttribute((const void*)out_kernel,    hipFuncAttributeMaxDynamicSharedMemorySize, 280576);

  prep_kernel<<<128, 256, 0, stream>>>(qkv_w, o1_w, o2_w, ws);
  qkv_kv_kernel<<<BATCH * 512, 256, 241664, stream>>>(x, kln_w, kln_b, vln_w, vln_b, ws);
  kvmat_kernel<<<512, 256, 0, stream>>>(ws);
  out_kernel<<<BATCH * 256, 256, 280576, stream>>>(x, out, ws);
}
